// STM32DfDecoder_14585708937407
// MI455X (gfx1250) — matrix operand order repaired and checked by reading, not compiled, not run on hardware
//
#include <hip/hip_runtime.h>
#include <hip/hip_bf16.h>
#include <math.h>

typedef __attribute__((ext_vector_type(16))) _Float16 v16h;
typedef __attribute__((ext_vector_type(8)))  float    v8f;

// ---------------------------------------------------------------------------
// Weight packing: fp32 -> f16 in WMMA B-fragment order.
// Packed layout: dst[(((nt*(K/32)+kc)*32 + lane)*16 + e], where
//   n = nt*16 + (lane&15),  k = kc*32 + (lane>>4)*16 + e
// so each lane's 16 halves for one K-chunk are one contiguous 32B load.
// modes: 0 = w_ih slice (3,512,512)[g][o][i] -> B[k=i][n=g*512+o]
//        1 = (O,I) dense transpose           -> B[k=i][n=o]
//        2 = grouped (8,64,64)[g][i][o] block-diagonal 512x512
//        3 = grouped (8,64,120)[g][i][o] block-diagonal 512x960
__global__ void pack_b(const float* __restrict__ src, _Float16* __restrict__ dst,
                       int N, int K, int mode, int total) {
  int tid = blockIdx.x * blockDim.x + threadIdx.x;
  if (tid >= total) return;
  int e    = tid & 15;
  int lane = (tid >> 4) & 31;
  int rest = tid >> 9;
  int K32  = K >> 5;
  int kc   = rest % K32;
  int nt   = rest / K32;
  int k = kc * 32 + (lane >> 4) * 16 + e;
  int n = nt * 16 + (lane & 15);
  float v = 0.f;
  if (mode == 0) {
    int g = n >> 9, o = n & 511;
    v = src[((size_t)(g * 512 + o)) * 512 + k];
  } else if (mode == 1) {
    v = src[(size_t)n * K + k];
  } else if (mode == 2) {
    int gn = n >> 6, o = n & 63, gk = k >> 6, i = k & 63;
    v = (gn == gk) ? src[(gn * 64 + i) * 64 + o] : 0.f;
  } else {
    int gn = n / 120, o = n % 120, gk = k >> 6, i = k & 63;
    v = (gn == gk) ? src[(gk * 64 + i) * 120 + o] : 0.f;
  }
  dst[tid] = (_Float16)v;
}

// ---------------------------------------------------------------------------
// WMMA GEMM: C[T x N] = epilogue( A[T x K] @ B + bias [+ addend] )
// One wave computes a 16x64 output strip (4 N-tiles): the A fragment
// (and its fp32->f16 conversion) is amortized over 4 v_wmma issues.
// mode: 0 = none, 1 = relu, 2 = tanh(acc) (+addend)
#define NT 4
__global__ void __launch_bounds__(128)
wmma_gemm(const float* __restrict__ A, const _Float16* __restrict__ Bp,
          const float* __restrict__ bias, const float* __restrict__ addend,
          float* __restrict__ C, int T, int N, int K, int mode) {
  int wave = threadIdx.x >> 5;
  int lane = threadIdx.x & 31;
  int nblk = N >> 6;                       // groups of 4 n-tiles
  int tot  = (T >> 4) * nblk;
  int tile = blockIdx.x * (blockDim.x >> 5) + wave;
  if (tile >= tot) return;                 // wave-uniform exit (EXEC all-1 for WMMA)
  int mt = tile / nblk, ntb = tile % nblk;
  int row = lane & 15, half = lane >> 4;
  int K32 = K >> 5;

  const float* Ar    = A + (size_t)(mt * 16 + row) * K + half * 16;
  const v16h*  Bbase = (const v16h*)Bp;

  v8f acc[NT];
#pragma unroll
  for (int j = 0; j < NT; ++j) acc[j] = (v8f){};

  for (int kc = 0; kc < K32; ++kc) {
    const float* Ab = Ar + kc * 32;
    float4 a0 = *(const float4*)(Ab);
    float4 a1 = *(const float4*)(Ab + 4);
    float4 a2 = *(const float4*)(Ab + 8);
    float4 a3 = *(const float4*)(Ab + 12);
    v16h af;
    af[0]  = (_Float16)a0.x; af[1]  = (_Float16)a0.y;
    af[2]  = (_Float16)a0.z; af[3]  = (_Float16)a0.w;
    af[4]  = (_Float16)a1.x; af[5]  = (_Float16)a1.y;
    af[6]  = (_Float16)a1.z; af[7]  = (_Float16)a1.w;
    af[8]  = (_Float16)a2.x; af[9]  = (_Float16)a2.y;
    af[10] = (_Float16)a2.z; af[11] = (_Float16)a2.w;
    af[12] = (_Float16)a3.x; af[13] = (_Float16)a3.y;
    af[14] = (_Float16)a3.z; af[15] = (_Float16)a3.w;
#pragma unroll
    for (int j = 0; j < NT; ++j) {
      v16h bf = Bbase[((size_t)(ntb * NT + j) * K32 + kc) * 32 + lane];
      acc[j] = __builtin_amdgcn_wmma_f32_16x16x32_f16(
          false, af, false, bf, (short)0, acc[j], false, false);
    }
  }

#pragma unroll
  for (int j = 0; j < NT; ++j) {
    int n = (ntb * NT + j) * 16 + row;
#pragma unroll
    for (int r = 0; r < 8; ++r) {
      int m = mt * 16 + r + half * 8;
      size_t idx = (size_t)m * N + n;
      float v = acc[j][r];
      if (mode == 2) {
        v = tanhf(v);
      } else {
        if (bias) v += bias[n];
        if (mode == 1) v = fmaxf(v, 0.f);
      }
      if (addend) v += addend[idx];
      C[idx] = v;
    }
  }
}

// ---------------------------------------------------------------------------
// Persistent-grid GRU recurrence, one layer. 32 WGs x 384 threads.
// WG wg owns h[wg*16 .. wg*16+16) for all 3 gates (48 outputs).
// Thread t: output lout = t%48 (g=lout>>4, jo=lout&15), K-chunk kc = t/48 (64 wide).
// 64 fp32 weights register-resident per thread; h broadcast via LDS;
// cross-WG step sync via double-buffered h + agent-scope arrival counter.
#define RWGS 32
#define RTHR 384
__global__ void __launch_bounds__(RTHR)
gru_rec(const float* __restrict__ ih,    // T x 1536 (gate-major: g*512+j)
        const float* __restrict__ whh,   // (3,512,512) layer slice [g][j][k]
        const float* __restrict__ bhh,   // 1536 layer slice
        float* __restrict__ hseq,        // T x 512 output
        float* __restrict__ hbuf,        // 2 x 512 double buffer (zeroed)
        unsigned int* __restrict__ cnt,  // arrival counter (zeroed)
        int T) {
  __shared__ float h_lds[512];
  __shared__ float part[RTHR];
  __shared__ float hhred[48];

  const int t  = threadIdx.x;
  const int wg = blockIdx.x;
  const int lout = t % 48, kc = t / 48;      // kc in 0..7
  const int g = lout >> 4, jo = lout & 15;
  const int jg = wg * 16 + jo;

  float wreg[64];
  const float* wp = whh + ((size_t)(g * 512 + jg)) * 512 + kc * 64;
#pragma unroll
  for (int i = 0; i < 64; ++i) wreg[i] = wp[i];

  for (int s = 0; s < T; ++s) {
    if (t == 0) {
      unsigned target = (unsigned)(RWGS * s);
      while (__hip_atomic_load(cnt, __ATOMIC_ACQUIRE, __HIP_MEMORY_SCOPE_AGENT) < target)
        __builtin_amdgcn_s_sleep(1);
    }
    __syncthreads();

    // broadcast h(s) into LDS (agent-scope loads: bypass stale WGP cache)
    const float* hsrc = hbuf + (s & 1) * 512;
    for (int i = t; i < 512; i += RTHR)
      h_lds[i] = __hip_atomic_load(&hsrc[i], __ATOMIC_RELAXED, __HIP_MEMORY_SCOPE_AGENT);
    __syncthreads();

    // partial dot: 64 register-resident weights x broadcast h
    float p = 0.f;
    const float* hk = h_lds + kc * 64;
#pragma unroll
    for (int i = 0; i < 64; ++i) p += wreg[i] * hk[i];
    part[t] = p;
    __syncthreads();

    if (t < 48) {
      float ssum = 0.f;
#pragma unroll
      for (int c = 0; c < 8; ++c) ssum += part[c * 48 + t];
      hhred[t] = ssum + bhh[(t >> 4) * 512 + wg * 16 + (t & 15)];
    }
    __syncthreads();

    if (t < 16) {
      const int j = wg * 16 + t;
      const float* iht = ih + (size_t)s * 1536;
      float r = 1.f / (1.f + expf(-(iht[j]        + hhred[t])));
      float z = 1.f / (1.f + expf(-(iht[512 + j]  + hhred[16 + t])));
      float n = tanhf(iht[1024 + j] + r * hhred[32 + t]);
      float hn = n + z * (h_lds[j] - n);
      __hip_atomic_store(&hbuf[((s + 1) & 1) * 512 + j], hn,
                         __ATOMIC_RELAXED, __HIP_MEMORY_SCOPE_AGENT);
      hseq[(size_t)s * 512 + j] = hn;
      __threadfence();
    }
    __syncthreads();
    if (t == 0)
      __hip_atomic_fetch_add(cnt, 1u, __ATOMIC_RELEASE, __HIP_MEMORY_SCOPE_AGENT);
  }
}

__global__ void init_zero(float* hbuf, unsigned int* cnt) {
  int t = blockIdx.x * blockDim.x + threadIdx.x;
  if (t < 1024) hbuf[t] = 0.f;
  if (t == 0) *cnt = 0u;
}

// alpha = sigmoid(c @ fc_a_w^T + b): one wave per timestep, coalesced
// lane-strided reads + wave32 shuffle reduction.
__global__ void alpha_k(const float* __restrict__ c, const float* __restrict__ w,
                        const float* __restrict__ b, float* __restrict__ out, int T) {
  int wave = threadIdx.x >> 5, lane = threadIdx.x & 31;
  int t0 = blockIdx.x * (blockDim.x >> 5) + wave;
  if (t0 >= T) return;
  const float* row = c + (size_t)t0 * 512;
  float acc = 0.f;
#pragma unroll
  for (int i = 0; i < 16; ++i) acc += row[lane + i * 32] * w[lane + i * 32];
#pragma unroll
  for (int m = 16; m >= 1; m >>= 1) acc += __shfl_xor(acc, m, 32);
  if (lane == 0) out[t0] = 1.f / (1.f + expf(-(acc + b[0])));
}

// causal KT=2 conv over T: writes base of d_out c-region (t,f,co) layout
__global__ void conv_k(const float* __restrict__ c0, const float* __restrict__ w,
                       const float* __restrict__ b, float* __restrict__ out, int T) {
  __shared__ float wl[1280];
  __shared__ float bl[10];
  for (int i = threadIdx.x; i < 1280; i += blockDim.x) wl[i] = w[i];
  if (threadIdx.x < 10) bl[threadIdx.x] = b[threadIdx.x];
  __syncthreads();
  int id = blockIdx.x * blockDim.x + threadIdx.x;
  if (id >= T * 96) return;
  int tt = id / 96, f = id % 96;
  float acc[10];
#pragma unroll
  for (int co = 0; co < 10; ++co) acc[co] = bl[co];
  for (int ci = 0; ci < 64; ++ci) {
    float x1 = c0[((size_t)ci * T + tt) * 96 + f];
    float x0 = (tt > 0) ? c0[((size_t)ci * T + tt - 1) * 96 + f] : 0.f;
#pragma unroll
    for (int co = 0; co < 10; ++co)
      acc[co] += x0 * wl[(co * 64 + ci) * 2 + 0] + x1 * wl[(co * 64 + ci) * 2 + 1];
  }
  float* o = out + (size_t)tt * 960 + f * 10;
#pragma unroll
  for (int co = 0; co < 10; ++co) o[co] = acc[co];
}

// ---------------------------------------------------------------------------
extern "C" void kernel_launch(void* const* d_in, const int* in_sizes, int n_in,
                              void* d_out, int out_size, void* d_ws, size_t ws_size,
                              hipStream_t stream) {
  const float* emb        = (const float*)d_in[0];
  const float* c0         = (const float*)d_in[1];
  const float* lin_in_w   = (const float*)d_in[2];
  const float* gru_w_ih   = (const float*)d_in[3];
  const float* gru_w_hh   = (const float*)d_in[4];
  const float* gru_b_ih   = (const float*)d_in[5];
  const float* gru_b_hh   = (const float*)d_in[6];
  const float* gru_skip_w = (const float*)d_in[7];
  const float* gru_skip_b = (const float*)d_in[8];
  const float* lin_out_w  = (const float*)d_in[9];
  const float* df_skip_w  = (const float*)d_in[10];
  const float* df_skip_b  = (const float*)d_in[11];
  const float* fc_a_w     = (const float*)d_in[12];
  const float* fc_a_b     = (const float*)d_in[13];
  const float* df_out_w   = (const float*)d_in[14];
  const float* convp_w    = (const float*)d_in[15];
  const float* convp_b    = (const float*)d_in[16];
  float* out = (float*)d_out;

  const int T = 8192, H = 512, HID = 512, N_IH = 1536, NDF = 960;

  char* ws = (char*)d_ws;
  size_t off = 0;
  auto alloc = [&](size_t bytes) -> char* {
    char* p = ws + off;
    off += (bytes + 255) & ~(size_t)255;
    return p;
  };
  float*     gru_in  = (float*)alloc((size_t)T * HID * 4);
  float*     ihbuf   = (float*)alloc((size_t)T * N_IH * 4);
  float*     hseq    = (float*)alloc((size_t)T * HID * 4);
  float*     cbuf    = (float*)alloc((size_t)T * HID * 4);
  float*     hbuf    = (float*)alloc(1024 * 4);
  unsigned*  cnt0    = (unsigned*)alloc(256);
  unsigned*  cnt1    = (unsigned*)alloc(256);
  _Float16*  Bih0    = (_Float16*)alloc((size_t)H * N_IH * 2);
  _Float16*  Bih1    = (_Float16*)alloc((size_t)H * N_IH * 2);
  _Float16*  Bskip   = (_Float16*)alloc((size_t)H * HID * 2);
  _Float16*  Bdfskip = (_Float16*)alloc((size_t)H * HID * 2);
  _Float16*  Blinin  = (_Float16*)alloc((size_t)H * HID * 2);
  _Float16*  Blinout = (_Float16*)alloc((size_t)HID * HID * 2);
  _Float16*  Bdfout  = (_Float16*)alloc((size_t)H * NDF * 2);
  float*     xbuf    = ihbuf;  // reuse: ih dead after second recurrence

  // --- pack weights (f32 -> WMMA-ready f16 fragments) ---
  int tot = H * N_IH;
  pack_b<<<(tot + 255) / 256, 256, 0, stream>>>(gru_w_ih,                 Bih0, N_IH, H, 0, tot);
  pack_b<<<(tot + 255) / 256, 256, 0, stream>>>(gru_w_ih + 3 * 512 * 512, Bih1, N_IH, H, 0, tot);
  tot = H * HID;
  pack_b<<<(tot + 255) / 256, 256, 0, stream>>>(gru_skip_w, Bskip,   HID, H,   1, tot);
  pack_b<<<(tot + 255) / 256, 256, 0, stream>>>(df_skip_w,  Bdfskip, HID, H,   1, tot);
  pack_b<<<(tot + 255) / 256, 256, 0, stream>>>(lin_in_w,   Blinin,  HID, H,   2, tot);
  pack_b<<<(tot + 255) / 256, 256, 0, stream>>>(lin_out_w,  Blinout, HID, HID, 2, tot);
  tot = H * NDF;
  pack_b<<<(tot + 255) / 256, 256, 0, stream>>>(df_out_w, Bdfout, NDF, H, 3, tot);

  auto gemm = [&](const float* A, const _Float16* Bp, const float* bias,
                  const float* add, float* C, int N, int K, int mode) {
    int tiles = (T / 16) * (N / 64);       // one wave per 16x64 strip
    wmma_gemm<<<(tiles + 3) / 4, 128, 0, stream>>>(A, Bp, bias, add, C, T, N, K, mode);
  };

  // 1) gru_in = relu(lin_in(emb))
  gemm(emb, Blinin, nullptr, nullptr, gru_in, HID, H, 1);
  // 2) ih layer0
  gemm(gru_in, Bih0, gru_b_ih, nullptr, ihbuf, N_IH, HID, 0);
  // 3) GRU layer 0
  init_zero<<<4, 256, 0, stream>>>(hbuf, cnt0);
  gru_rec<<<RWGS, RTHR, 0, stream>>>(ihbuf, gru_w_hh, gru_b_hh, hseq, hbuf, cnt0, T);
  // 4) ih layer1
  gemm(hseq, Bih1, gru_b_ih + 1536, nullptr, ihbuf, N_IH, HID, 0);
  // 5) GRU layer 1 (overwrites hseq in place)
  init_zero<<<4, 256, 0, stream>>>(hbuf, cnt1);
  gru_rec<<<RWGS, RTHR, 0, stream>>>(ihbuf, gru_w_hh + 3 * 512 * 512,
                                     gru_b_hh + 1536, hseq, hbuf, cnt1, T);
  // 6) x = hseq + gru_in @ gru_skip^T + b
  gemm(gru_in, Bskip, gru_skip_b, hseq, xbuf, HID, HID, 0);
  // 7) c_pre = relu(lin_out(x))
  gemm(xbuf, Blinout, nullptr, nullptr, cbuf, HID, HID, 1);
  // 8) c = c_pre + emb @ df_skip^T + b   (in-place add into cbuf)
  gemm(emb, Bdfskip, df_skip_b, cbuf, cbuf, HID, H, 0);
  // 9) alpha (one wave per timestep, 8 waves per block)
  alpha_k<<<(T + 7) / 8, 256, 0, stream>>>(cbuf, fc_a_w, fc_a_b,
                                           out + (size_t)T * NDF, T);
  // 10) conv path writes d_out base
  conv_k<<<(T * 96 + 255) / 256, 256, 0, stream>>>(c0, convp_w, convp_b, out, T);
  // 11) d_out += tanh(df_out(c))
  gemm(cbuf, Bdfout, nullptr, out, out, NDF, H, 2);
}
